// QAConv_77438260347348
// MI455X (gfx1250) — compile-verified, matrix operand order checked by reading, not run on hardware
//
#include <hip/hip_runtime.h>
#include <stdint.h>

// Problem constants (from reference): P=G=64, C=128, H*W=192
#define P_N   64
#define G_N   64
#define C_N   128
#define HW_N  192
#define PAIRS (P_N * G_N)
#define EPSF  1e-5f

typedef __attribute__((ext_vector_type(16))) __bf16 v16bf;
typedef __attribute__((ext_vector_type(8)))  float  v8f;
typedef unsigned int v4u __attribute__((ext_vector_type(4)));
typedef int          v8i __attribute__((ext_vector_type(8)));
typedef int          v4i __attribute__((ext_vector_type(4)));

// ---------- helpers ----------
__device__ __forceinline__ unsigned short f2bf(float f) {
    unsigned b = __float_as_uint(f);
    b += 0x7fffu + ((b >> 16) & 1u);   // round-to-nearest-even
    return (unsigned short)(b >> 16);
}
// order-preserving float <-> uint map so integer atomicMax works in LDS
__device__ __forceinline__ unsigned mapOrd(float f) {
    unsigned b = __float_as_uint(f);
    return (b & 0x80000000u) ? ~b : (b | 0x80000000u);
}
__device__ __forceinline__ float unmapOrd(unsigned u) {
    unsigned b = (u & 0x80000000u) ? (u & 0x7fffffffu) : ~u;
    return __uint_as_float(b);
}
__device__ __forceinline__ float blockReduce256(float x, float* red, int tid) {
    red[tid] = x; __syncthreads();
    #pragma unroll
    for (int s = 128; s > 0; s >>= 1) {
        if (tid < s) red[tid] += red[tid + s];
        __syncthreads();
    }
    float r = red[0]; __syncthreads();
    return r;
}

#if __has_builtin(__builtin_amdgcn_tensor_load_to_lds)
#define USE_TDM 1
// TDM 2D tile load: rows = HW_N(192), row width = 32 bf16 (64B), global row
// stride = C_N elements; destination LDS is packed contiguously [row][32].
__device__ __forceinline__ void tdm_load_2d(const unsigned short* gsrc, unsigned ldsOff) {
    unsigned long long ga = (unsigned long long)(uintptr_t)gsrc;
    v4u g0;
    g0.x = 1u;                                                // count=1, user D#
    g0.y = ldsOff;                                            // lds_addr (bytes)
    g0.z = (unsigned)(ga & 0xffffffffu);                      // global_addr[31:0]
    g0.w = (unsigned)((ga >> 32) & 0x01ffffffu) | (2u << 30); // addr[56:32] | type=2
    v8i g1;
    g1[0] = (1 << 16);        // data_size=1 -> 2 bytes; wg_mask=0 (not in cluster)
    g1[1] = (C_N << 16);      // tensor_dim0 = 128  (bits[79:48] low half)
    g1[2] = (HW_N << 16);     // tensor_dim0 hi=0 | tensor_dim1 = 192 (low 16)
    g1[3] = (32 << 16);       // tensor_dim1 hi=0 | tile_dim0 = 32
    g1[4] = HW_N;             // tile_dim1 = 192, tile_dim2 = 0 (2D)
    g1[5] = C_N;              // tensor_dim0_stride = 128 (48-bit, low word)
    g1[6] = 0;
    g1[7] = 0;
    v4i z4 = {0, 0, 0, 0};
#if __clang_major__ >= 23
    v8i z8 = {0, 0, 0, 0, 0, 0, 0, 0};
    __builtin_amdgcn_tensor_load_to_lds(g0, g1, z4, z4, z8, 0);
#else
    __builtin_amdgcn_tensor_load_to_lds(g0, g1, z4, z4, 0);
#endif
}
#else
#define USE_TDM 0
#endif

// ---------- kernel 0: (B, C, HW) f32 -> (B, HW, C) bf16 (K-contiguous) ----------
__global__ void qaconv_pack(const float* __restrict__ src,
                            unsigned short* __restrict__ dst, int total) {
    int i = blockIdx.x * blockDim.x + threadIdx.x;
    if (i >= total) return;
    int b   = i / (C_N * HW_N);
    int rem = i - b * (C_N * HW_N);
    int c   = rem / HW_N;
    int x   = rem - c * HW_N;
    dst[((size_t)b * HW_N + x) * C_N + c] = f2bf(src[i]);
}

// ---------- kernel 1: per-pair 192x192x128 WMMA GEMM + row/col max pooling ----------
// block = 256 threads (8 waves), one block per (probe, gallery) pair.
// Tile map: wave owns tile-rows 3*(wave&3)..+3  x  tile-cols 6*(wave>>2)..+6.
__global__ void __launch_bounds__(256)
qaconv_pair(const unsigned short* __restrict__ pPack,   // (P, HW, C) bf16
            const unsigned short* __restrict__ gPack,   // (G, HW, C) bf16
            const float* __restrict__ fcw,              // (HW,)
            float* __restrict__ pairWS, float* __restrict__ pairS1,
            float* __restrict__ pairS2) {
    __shared__ __align__(16) unsigned short ldsA[2][HW_N * 32]; // gallery K-chunks [r][k]
    __shared__ __align__(16) unsigned short ldsB[2][HW_N * 32]; // probe   K-chunks [s][k]
    __shared__ unsigned rowmaxU[HW_N];   // max over s, per r  (= reference max_s)
    __shared__ unsigned colmaxU[HW_N];   // max over r, per s  (= reference max_r)
    __shared__ float redW[8], redS1[8], redS2[8];

    const int tid  = threadIdx.x;
    const int pair = blockIdx.x;
    const int pi   = pair >> 6;
    const int gi   = pair & 63;
    const unsigned short* pBase = pPack + (size_t)pi * HW_N * C_N;
    const unsigned short* gBase = gPack + (size_t)gi * HW_N * C_N;

    if (tid < HW_N) { rowmaxU[tid] = 0u; colmaxU[tid] = 0u; }

    const int wave   = tid >> 5;
    const int lane   = tid & 31;
    const int mIdx   = lane & 15;   // A row / B col / C col within tile
    const int hi     = lane >> 4;   // half-wave -> K split (A/B), M split (C)
    const int wave_r = wave & 3;    // tile-rows 3*wave_r .. +3
    const int wave_c = wave >> 2;   // tile-cols 6*wave_c .. +6

    v8f acc[3][6] = {};

#if USE_TDM
    const unsigned offA0 = (unsigned)(uintptr_t)&ldsA[0][0];
    const unsigned offA1 = (unsigned)(uintptr_t)&ldsA[1][0];
    const unsigned offB0 = (unsigned)(uintptr_t)&ldsB[0][0];
    const unsigned offB1 = (unsigned)(uintptr_t)&ldsB[1][0];
    if (wave == 0) {                 // TDM is wave-level: single issuing wave
        tdm_load_2d(gBase, offA0);
        tdm_load_2d(pBase, offB0);
    }
#endif

    for (int kc = 0; kc < 4; ++kc) {
        const int buf = kc & 1;
#if USE_TDM
        if (wave == 0) {
            if (kc < 3) {            // prefetch next K-chunk into the other buffer
                tdm_load_2d(gBase + (kc + 1) * 32, buf ? offA0 : offA1);
                tdm_load_2d(pBase + (kc + 1) * 32, buf ? offB0 : offB1);
                __builtin_amdgcn_s_wait_tensorcnt((short)2);  // current chunk done
            } else {
                __builtin_amdgcn_s_wait_tensorcnt((short)0);
            }
        }
        __syncthreads();             // release all waves onto completed chunk
#else
        __syncthreads();
        for (int e = tid; e < 768; e += 256) {   // 768 x 16B per matrix chunk
            int row = e >> 2, q = e & 3;
            ((uint4*)ldsA[buf])[e] = *(const uint4*)(gBase + row * C_N + kc * 32 + q * 8);
            ((uint4*)ldsB[buf])[e] = *(const uint4*)(pBase + row * C_N + kc * 32 + q * 8);
        }
        __syncthreads();
#endif
        const unsigned short* A = ldsA[buf];
        const unsigned short* B = ldsB[buf];
        const uint4* brow = (const uint4*)(B + (wave_c * 96 + mIdx) * 32);
        #pragma unroll
        for (int rr = 0; rr < 3; ++rr) {
            union { uint4 q[2]; v16bf v; } fa;
            const uint4* ar = (const uint4*)(A + ((wave_r * 3 + rr) * 16 + mIdx) * 32);
            fa.q[0] = ar[hi]; fa.q[1] = ar[2 + hi];          // regs0-3: K[hi*8..], regs4-7: K[16+hi*8..]
            #pragma unroll
            for (int j = 0; j < 6; ++j) {
                union { uint4 q[2]; v16bf v; } fb;
                fb.q[0] = brow[j * 64 + hi];                 // j*16 rows * 4 uint4/row
                fb.q[1] = brow[j * 64 + 2 + hi];
                acc[rr][j] = __builtin_amdgcn_wmma_f32_16x16x32_bf16(
                    false, fa.v, false, fb.v, (short)0, acc[rr][j], false, false);
            }
        }
        __syncthreads();             // chunk fully consumed before TDM overwrites it
    }

    // ---- pooling. C-fragment layout: lane col n = mIdx, reg v -> row M = hi*8+v ----
    // Column max (max over r): combine the wave's 3 row-bands elementwise first,
    // then one in-lane tree + one half-wave combine per column tile.
    #pragma unroll
    for (int j = 0; j < 6; ++j) {
        const int ts = wave_c * 6 + j;
        float cm = fmaxf(fmaxf(acc[0][j][0], acc[1][j][0]), acc[2][j][0]);
        #pragma unroll
        for (int v = 1; v < 8; ++v)
            cm = fmaxf(cm, fmaxf(fmaxf(acc[0][j][v], acc[1][j][v]), acc[2][j][v]));
        cm = fmaxf(cm, __shfl_xor(cm, 16));
        if (hi == 0) atomicMax(&colmaxU[ts * 16 + mIdx], mapOrd(cm));
    }
    // Row max (max over s): combine the wave's 6 column-tiles elementwise first,
    // then one 16-lane shuffle tree per register slot.
    #pragma unroll
    for (int rr = 0; rr < 3; ++rr) {
        const int tr = wave_r * 3 + rr;
        #pragma unroll
        for (int v = 0; v < 8; ++v) {
            float rv = acc[rr][0][v];
            #pragma unroll
            for (int j = 1; j < 6; ++j) rv = fmaxf(rv, acc[rr][j][v]);
            rv = fmaxf(rv, __shfl_xor(rv, 1));
            rv = fmaxf(rv, __shfl_xor(rv, 2));
            rv = fmaxf(rv, __shfl_xor(rv, 4));
            rv = fmaxf(rv, __shfl_xor(rv, 8));
            if (mIdx == v) atomicMax(&rowmaxU[tr * 16 + hi * 8 + v], mapOrd(rv));
        }
    }
    __syncthreads();

    // ---- reduce the 384 maxima -> fc_w-weighted sum, sum, sum of squares ----
    float ws = 0.f, s1 = 0.f, s2 = 0.f;
    for (int idx = tid; idx < 2 * HW_N; idx += 256) {
        int   col = (idx < HW_N) ? idx : idx - HW_N;
        float v   = (idx < HW_N) ? unmapOrd(rowmaxU[idx]) : unmapOrd(colmaxU[col]);
        ws += v * fcw[col];
        s1 += v;
        s2 += v * v;
    }
    #pragma unroll
    for (int m = 1; m < 32; m <<= 1) {       // fused wave-level reduction
        ws += __shfl_xor(ws, m);
        s1 += __shfl_xor(s1, m);
        s2 += __shfl_xor(s2, m);
    }
    if (lane == 0) { redW[wave] = ws; redS1[wave] = s1; redS2[wave] = s2; }
    __syncthreads();
    if (tid == 0) {
        float W = 0.f, a = 0.f, b = 0.f;
        #pragma unroll
        for (int w8 = 0; w8 < 8; ++w8) { W += redW[w8]; a += redS1[w8]; b += redS2[w8]; }
        pairWS[pair] = W; pairS1[pair] = a; pairS2[pair] = b;
    }
}

// ---------- kernel 2: both BatchNorms + output (single block, deterministic) ----------
__global__ void __launch_bounds__(256)
qaconv_finalize(const float* __restrict__ pairWS, const float* __restrict__ pairS1,
                const float* __restrict__ pairS2,
                const float* __restrict__ bn_g, const float* __restrict__ bn_b,
                const float* __restrict__ fcw,  const float* __restrict__ fcb,
                const float* __restrict__ lg,   const float* __restrict__ lb,
                float* __restrict__ out) {
    __shared__ float red[256];
    __shared__ float scBuf[PAIRS];
    const int tid = threadIdx.x;

    float w = 0.f;
    for (int i = tid; i < HW_N; i += 256) w += fcw[i];
    float Wtot = blockReduce256(w, red, tid);

    float s1 = 0.f, s2 = 0.f;
    for (int i = tid; i < PAIRS; i += 256) { s1 += pairS1[i]; s2 += pairS2[i]; }
    float S1 = blockReduce256(s1, red, tid);
    float S2 = blockReduce256(s2, red, tid);

    const float N1 = (float)PAIRS * (float)(2 * HW_N);
    float mu  = S1 / N1;
    float var = S2 / N1 - mu * mu;
    float a   = rsqrtf(var + EPSF) * bn_g[0];
    float C0  = -2.f * a * mu * Wtot + 2.f * bn_b[0] * Wtot + 2.f * fcb[0];

    float t1 = 0.f, t2 = 0.f;
    for (int i = tid; i < PAIRS; i += 256) {
        float sc = a * pairWS[i] + C0;
        scBuf[i] = sc;
        t1 += sc; t2 += sc * sc;
    }
    float T1 = blockReduce256(t1, red, tid);
    float T2 = blockReduce256(t2, red, tid);
    float mu2  = T1 / (float)PAIRS;
    float var2 = T2 / (float)PAIRS - mu2 * mu2;
    float inv2 = rsqrtf(var2 + EPSF) * lg[0];
    float beta2 = lb[0];
    for (int i = tid; i < PAIRS; i += 256)
        out[i] = (scBuf[i] - mu2) * inv2 + beta2;
}

// ---------- launch ----------
extern "C" void kernel_launch(void* const* d_in, const int* in_sizes, int n_in,
                              void* d_out, int out_size, void* d_ws, size_t ws_size,
                              hipStream_t stream) {
    const float* prob = (const float*)d_in[0];
    const float* gal  = (const float*)d_in[1];
    const float* bn_g = (const float*)d_in[2];
    const float* bn_b = (const float*)d_in[3];
    const float* fcw  = (const float*)d_in[4];
    const float* fcb  = (const float*)d_in[5];
    const float* lg   = (const float*)d_in[6];
    const float* lb   = (const float*)d_in[7];
    float* out = (float*)d_out;

    unsigned short* pPack = (unsigned short*)d_ws;
    unsigned short* gPack = pPack + (size_t)P_N * HW_N * C_N;
    float* pairWS = (float*)(gPack + (size_t)G_N * HW_N * C_N);
    float* pairS1 = pairWS + PAIRS;
    float* pairS2 = pairS1 + PAIRS;

    const int total = P_N * C_N * HW_N;
    qaconv_pack<<<(total + 255) / 256, 256, 0, stream>>>(prob, pPack, total);
    qaconv_pack<<<(total + 255) / 256, 256, 0, stream>>>(gal,  gPack, total);
    qaconv_pair<<<PAIRS, 256, 0, stream>>>(pPack, gPack, fcw, pairWS, pairS1, pairS2);
    qaconv_finalize<<<1, 256, 0, stream>>>(pairWS, pairS1, pairS2,
                                           bn_g, bn_b, fcw, fcb, lg, lb, out);
}
